// fusionblock_72567767433473
// MI455X (gfx1250) — matrix operand order repaired and checked by reading, not compiled, not run on hardware
//
#include <hip/hip_runtime.h>
#include <hip/hip_bf16.h>
#include <math.h>

// ---------------- constants ----------------
#define NPTS 100000
#define CH   256
#define KNB  16
#define NLAY 2
#define NEGS 0.1f
#define EPSL 1e-5f

// ---------------- CDNA5 WMMA types ----------------
typedef __bf16        v16bf __attribute__((ext_vector_type(16)));
typedef float         v8f   __attribute__((ext_vector_type(8)));
typedef unsigned int  v8u   __attribute__((ext_vector_type(8)));

static __device__ __forceinline__ v8f wmma_bf16(v16bf a, v16bf b, v8f c) {
    // v_wmma_f32_16x16x32_bf16  (8-arg form: neg_a, A, neg_b, B, c_mod, C, reuse_a, reuse_b)
    return __builtin_amdgcn_wmma_f32_16x16x32_bf16(false, a, false, b, (short)0, c, false, false);
}

// float -> bf16 bits, round-to-nearest-even
static __device__ __forceinline__ unsigned short f2bf(float f) {
    unsigned u = __builtin_bit_cast(unsigned, f);
    u += 0x7FFFu + ((u >> 16) & 1u);
    return (unsigned short)(u >> 16);
}
static __device__ __forceinline__ unsigned pack2(float lo, float hi) {
    return (unsigned)f2bf(lo) | ((unsigned)f2bf(hi) << 16);
}

// ---- A fragment (16x32 bf16, M x K) built from an fp32 LDS tile (row-major, stride S floats)
// 16-bit A layout: lane<16: M=lane, VGPR0-3 K=0..7, VGPR4-7 K=16..23 ; lane>=16: K+8 offset
static __device__ __forceinline__ v16bf a_frag_lds_f32(const float* t, int S, int k0, int lane) {
    int half = (lane >> 4) & 1, m = lane & 15;
    const float* row = t + m * S;
    v8u u;
#pragma unroll
    for (int i = 0; i < 8; ++i) {
        int kb = k0 + ((i >> 2) << 4) + (half << 3) + ((i & 3) << 1);
        u[i] = pack2(row[kb], row[kb + 1]);
    }
    return __builtin_bit_cast(v16bf, u);
}
// ---- A fragment from global bf16 (row-major, Kdim stride)
static __device__ __forceinline__ v16bf a_frag_glb_bf(const unsigned short* a, int Kdim, int k0, int lane) {
    int half = (lane >> 4) & 1, m = lane & 15;
    const unsigned* p0 = (const unsigned*)(a + (size_t)m * Kdim + k0 + (half << 3));
    const unsigned* p1 = (const unsigned*)(a + (size_t)m * Kdim + k0 + 16 + (half << 3));
    v8u u;
    u[0] = p0[0]; u[1] = p0[1]; u[2] = p0[2]; u[3] = p0[3];
    u[4] = p1[0]; u[5] = p1[1]; u[6] = p1[2]; u[7] = p1[3];
    return __builtin_bit_cast(v16bf, u);
}
// ---- A fragment from LDS bf16 (row-major ushort, stride S elems)
static __device__ __forceinline__ v16bf a_frag_lds_bf(const unsigned short* t, int S, int k0, int lane) {
    int half = (lane >> 4) & 1, m = lane & 15;
    const unsigned short* row = t + m * S;
    v8u u;
#pragma unroll
    for (int i = 0; i < 8; ++i) {
        int kb = k0 + ((i >> 2) << 4) + (half << 3) + ((i & 3) << 1);
        u[i] = *(const unsigned*)(row + kb);
    }
    return __builtin_bit_cast(v16bf, u);
}
// ---- B fragment (32x16 bf16, K x N) from pre-transposed global weights Bt[n][k] (k-contiguous)
// B layout: lane<16: N=lane, VGPR0-3 K=0..7, VGPR4-7 K=16..23 ; lane>=16: K+8 offset
static __device__ __forceinline__ v16bf b_frag_glb(const unsigned short* bt, int Kdim, int n0, int k0, int lane) {
    int half = (lane >> 4) & 1, nc = n0 + (lane & 15);
    const unsigned* p = (const unsigned*)(bt + (size_t)nc * Kdim + k0 + (half << 3));
    v8u u;
#pragma unroll
    for (int i = 0; i < 8; ++i) u[i] = p[((i >> 2) << 3) + (i & 3)];
    return __builtin_bit_cast(v16bf, u);
}

static __device__ __forceinline__ float angle3(float ax, float ay, float az,
                                               float bx, float by, float bz) {
    float cx = ay * bz - az * by, cy = az * bx - ax * bz, cz = ax * by - ay * bx;
    float cr = sqrtf(cx * cx + cy * cy + cz * cz);
    float dt = ax * bx + ay * by + az * bz;
    return atan2f(cr, dt);
}

// 8-step K loop, double-buffered B + dual accumulators (breaks WMMA RAW chain).
static __device__ __forceinline__ v8f gemm_k8(const v16bf* af, const unsigned short* bt,
                                              int Kdim, int n0, int lane) {
    v8f acc0 = {0.f, 0.f, 0.f, 0.f, 0.f, 0.f, 0.f, 0.f};
    v8f acc1 = acc0;
    v16bf bA = b_frag_glb(bt, Kdim, n0, 0, lane);
    v16bf bB = b_frag_glb(bt, Kdim, n0, 32, lane);
#pragma unroll
    for (int kk = 0; kk < 8; kk += 2) {
        v16bf bC, bD;
        if (kk + 2 < 8) bC = b_frag_glb(bt, Kdim, n0, (kk + 2) * 32, lane);
        if (kk + 3 < 8) bD = b_frag_glb(bt, Kdim, n0, (kk + 3) * 32, lane);
        acc0 = wmma_bf16(af[kk], bA, acc0);
        acc1 = wmma_bf16(af[kk + 1], bB, acc1);
        bA = bC; bB = bD;
    }
    return acc0 + acc1;
}

// ---------------- weight prep kernels ----------------
__global__ void k_transpose_bf16(const float* __restrict__ src, unsigned short* __restrict__ dst,
                                 int R, int Cc) {
    int i = blockIdx.x * blockDim.x + threadIdx.x;
    int tot = R * Cc;
    if (i >= tot) return;
    int r = i / Cc, c = i - r * Cc;
    dst[(size_t)c * R + r] = f2bf(src[i]);
}
__global__ void k_convert_bf16(const float* __restrict__ src, unsigned short* __restrict__ dst, int n) {
    int i = blockIdx.x * blockDim.x + threadIdx.x;
    if (i < n) dst[i] = f2bf(src[i]);
}

// ---------------- Kernel A: geometry + embed MLP + 256x256 WMMA + gather + conv + maxpool ----------------
// One wave per node (M=16 neighbors). 8 waves / block.
#define HTS 260
#define G1S 68
#define WAVE_F (16 * HTS + 16 * G1S + 16 * 8)              // floats per wave
#define SMEM_KA (8 * WAVE_F * 4 + 8 * 16 * 4)              // + idx ints

__global__ void __launch_bounds__(256, 1)
k_node_embed(const float* __restrict__ p, const float* __restrict__ nrm,
             const int* __restrict__ idx, const float* __restrict__ xsrc,
             const float* __restrict__ rr,
             const float* __restrict__ emb_w1, const float* __restrict__ emb_g1,
             const float* __restrict__ emb_b1,
             const unsigned short* __restrict__ w2t,
             const unsigned short* __restrict__ convb,
             unsigned short* __restrict__ pooledb, int layer) {
    extern __shared__ char smem[];
    const int tid = threadIdx.x, wave = tid >> 5, lane = tid & 31;
    const int node = blockIdx.x * 8 + wave;

    float* ht  = (float*)smem + (size_t)wave * WAVE_F;   // 16 x HTS   (hemb, later g)
    float* g1t = ht + 16 * HTS;                          // 16 x G1S   (conv out)
    float* f5  = g1t + 16 * G1S;                         // 16 x 8
    int*   ids = (int*)(smem + 8 * WAVE_F * 4) + wave * 16;

    // ---- 1. geometry: PPF + sphere for the 16 neighbors (lanes 0..15)
    if (lane < 16) {
        int j = idx[(size_t)node * (2 * KNB) + lane];
        ids[lane] = j;
        float px = p[node * 3 + 0], py = p[node * 3 + 1], pz = p[node * 3 + 2];
        float qx = p[j * 3 + 0],    qy = p[j * 3 + 1],    qz = p[j * 3 + 2];
        float dx = qx - px, dy = qy - py, dz = qz - pz;
        float dist = sqrtf(dx * dx + dy * dy + dz * dz);
        float n1x = nrm[node * 3 + 0], n1y = nrm[node * 3 + 1], n1z = nrm[node * 3 + 2];
        float ngx = nrm[j * 3 + 0],    ngy = nrm[j * 3 + 1],    ngz = nrm[j * 3 + 2];
        f5[lane * 8 + 0] = angle3(n1x, n1y, n1z, dx, dy, dz);
        f5[lane * 8 + 1] = angle3(ngx, ngy, ngz, dx, dy, dz);
        f5[lane * 8 + 2] = angle3(n1x, n1y, n1z, ngx, ngy, ngz);
        f5[lane * 8 + 3] = dist;
        f5[lane * 8 + 4] = dist / rr[0];
    }
    __builtin_amdgcn_wave_barrier();

    // ---- 2. hemb = f5 @ emb_w1 (16x5 @ 5x256), LayerNorm + leaky; lane handles (m = lane/2, 128 cols)
    {
        const float* w1 = emb_w1 + (size_t)layer * 5 * CH;
        const float* g1 = emb_g1 + (size_t)layer * CH;
        const float* b1 = emb_b1 + (size_t)layer * CH;
        int m = lane >> 1, c0 = (lane & 1) * 128;
        float f0 = f5[m * 8 + 0], f1 = f5[m * 8 + 1], f2 = f5[m * 8 + 2],
              f3 = f5[m * 8 + 3], f4 = f5[m * 8 + 4];
        float s = 0.f, q = 0.f;
        for (int c = 0; c < 128; ++c) {
            int cc = c0 + c;
            float h = f0 * w1[0 * CH + cc] + f1 * w1[1 * CH + cc] + f2 * w1[2 * CH + cc]
                    + f3 * w1[3 * CH + cc] + f4 * w1[4 * CH + cc];
            ht[m * HTS + cc] = h;
            s += h; q += h * h;
        }
        s += __shfl_xor(s, 1, 32);
        q += __shfl_xor(q, 1, 32);
        float mean = s * (1.0f / CH);
        float var  = q * (1.0f / CH) - mean * mean;
        float rq   = rsqrtf(var + EPSL);
        for (int c = 0; c < 128; ++c) {
            int cc = c0 + c;
            float v = (ht[m * HTS + cc] - mean) * rq * g1[cc] + b1[cc];
            ht[m * HTS + cc] = (v > 0.f) ? v : NEGS * v;
        }
    }
    __builtin_amdgcn_wave_barrier();

    // ---- 3a. pos = hemb @ emb_w2 (16x256 @ 256x256) via WMMA; store pos tiles into ht
    {
        const unsigned short* w2l = w2t + (size_t)layer * CH * CH;
        __builtin_prefetch(w2l, 0, 1);
        v16bf af[8];
#pragma unroll
        for (int kk = 0; kk < 8; ++kk) af[kk] = a_frag_lds_f32(ht, HTS, kk * 32, lane);
        int half = lane >> 4, nl = lane & 15;
        for (int nt = 0; nt < 16; ++nt) {
            v8f acc = gemm_k8(af, w2l, CH, nt * 16, lane);
            int col = nt * 16 + nl;
#pragma unroll
            for (int i = 0; i < 8; ++i) ht[(i + half * 8) * HTS + col] = acc[i];
        }
    }
    __builtin_amdgcn_wave_barrier();

    // ---- 3b. g = pos + x[idx]: coalesced float4 gather-add, one neighbor row at a time
    {
        for (int m = 0; m < 16; ++m) {
            const float4* xr = (const float4*)(xsrc + (size_t)ids[m] * CH);
            float4 v0 = xr[lane];        // cols 4*lane   .. 4*lane+3
            float4 v1 = xr[lane + 32];   // cols 128+4*lane ..
            float* hrow = ht + m * HTS + lane * 4;
            hrow[0] += v0.x; hrow[1] += v0.y; hrow[2] += v0.z; hrow[3] += v0.w;
            hrow[128] += v1.x; hrow[129] += v1.y; hrow[130] += v1.z; hrow[131] += v1.w;
        }
    }
    __builtin_amdgcn_wave_barrier();

    // ---- 4. conv on first 64 channels: g1t = g[:, :64] @ conv_w^T (64x64) via WMMA
    {
        const unsigned short* cvl = convb + (size_t)layer * 64 * 64;
        v16bf ca0 = a_frag_lds_f32(ht, HTS, 0, lane);
        v16bf ca1 = a_frag_lds_f32(ht, HTS, 32, lane);
        int half = lane >> 4, nl = lane & 15;
#pragma unroll
        for (int nt = 0; nt < 4; ++nt) {
            v16bf cb0 = b_frag_glb(cvl, 64, nt * 16, 0, lane);
            v16bf cb1 = b_frag_glb(cvl, 64, nt * 16, 32, lane);
            v8f acc = {0.f, 0.f, 0.f, 0.f, 0.f, 0.f, 0.f, 0.f};
            acc = wmma_bf16(ca0, cb0, acc);
            acc = wmma_bf16(ca1, cb1, acc);
            int col = nt * 16 + nl;
#pragma unroll
            for (int i = 0; i < 8; ++i) g1t[(i + half * 8) * G1S + col] = acc[i];
        }
    }
    __builtin_amdgcn_wave_barrier();

    // ---- 5. max-pool over the 16 neighbors -> pooled (bf16), lane handles 8 channels
    {
        unsigned pk[4];
#pragma unroll
        for (int t = 0; t < 4; ++t) {
            float mx0 = -3.4e38f, mx1 = -3.4e38f;
            int c = lane * 8 + t * 2;
#pragma unroll
            for (int m = 0; m < 16; ++m) {
                float v0 = (c < 64)     ? g1t[m * G1S + c]     : ht[m * HTS + c];
                float v1 = (c + 1 < 64) ? g1t[m * G1S + c + 1] : ht[m * HTS + c + 1];
                mx0 = fmaxf(mx0, v0);
                mx1 = fmaxf(mx1, v1);
            }
            pk[t] = pack2(mx0, mx1);
        }
        uint4 w; w.x = pk[0]; w.y = pk[1]; w.z = pk[2]; w.w = pk[3];
        *reinterpret_cast<uint4*>(pooledb + (size_t)node * CH + lane * 8) = w;
    }
}

// ---------------- Kernel B: pointwise MLP (256 -> 1024 -> 256) with LNs, residual ----------------
#define S1 1032
#define S2 264
#define PW_H1   (16 * S1 * 4)
#define PW_H1B  (16 * S1 * 2)
#define PW_H2   (16 * S2 * 4)
#define SMEM_PW (PW_H1 + PW_H1B + PW_H2 + 2 * 256 * 4 + 2 * 16 * 4)

__global__ void __launch_bounds__(256, 1)
k_pointwise(const unsigned short* __restrict__ pooledb, const float* __restrict__ xcur,
            const unsigned short* __restrict__ w1t,
            const float* __restrict__ pg1, const float* __restrict__ pb1,
            const unsigned short* __restrict__ w2pt,
            const float* __restrict__ pg2, const float* __restrict__ pb2,
            float* __restrict__ hout, unsigned short* __restrict__ houtb, int layer) {
    extern __shared__ char smem[];
    float*          h1   = (float*)smem;                                  // 16 x S1 f32
    unsigned short* h1b  = (unsigned short*)(smem + PW_H1);               // 16 x S1 bf16
    float*          h2   = (float*)(smem + PW_H1 + PW_H1B);               // 16 x S2 f32
    float*          reds = (float*)(smem + PW_H1 + PW_H1B + PW_H2);       // 256
    float*          redq = reds + 256;                                    // 256
    float*          stm  = redq + 256;                                    // 16
    float*          stv  = stm + 16;                                      // 16

    const int tid = threadIdx.x, wave = tid >> 5, lane = tid & 31;
    const int row0 = blockIdx.x * 16;
    const int half = lane >> 4, nl = lane & 15;

    const unsigned short* w1l = w1t  + (size_t)layer * 4 * CH * CH;       // [1024][256]
    const unsigned short* w2l = w2pt + (size_t)layer * 4 * CH * CH;       // [256][1024]
    const float* g1 = pg1 + (size_t)layer * 4 * CH;
    const float* b1 = pb1 + (size_t)layer * 4 * CH;
    const float* g2 = pg2 + (size_t)layer * CH;
    const float* b2 = pb2 + (size_t)layer * CH;
    __builtin_prefetch(w1l, 0, 1);

    // ---- GEMM1: (16x256) @ (256x1024); wave w owns 8 output tiles
    {
        const unsigned short* ab = pooledb + (size_t)row0 * CH;
        v16bf af[8];
#pragma unroll
        for (int kk = 0; kk < 8; ++kk) af[kk] = a_frag_glb_bf(ab, CH, kk * 32, lane);
        for (int t = 0; t < 8; ++t) {
            int nt = wave * 8 + t;
            v8f acc = gemm_k8(af, w1l, CH, nt * 16, lane);
            int col = nt * 16 + nl;
#pragma unroll
            for (int i = 0; i < 8; ++i) h1[(i + half * 8) * S1 + col] = acc[i];
        }
    }
    __syncthreads();

    // ---- LN(1024) + leaky -> bf16
    {
        int r = tid >> 4, part = tid & 15;
        float s = 0.f, q = 0.f;
        for (int c = part * 64; c < part * 64 + 64; ++c) {
            float v = h1[r * S1 + c]; s += v; q += v * v;
        }
        reds[r * 16 + part] = s; redq[r * 16 + part] = q;
        __syncthreads();
        if (part == 0) {
            float S = 0.f, Q = 0.f;
            for (int k = 0; k < 16; ++k) { S += reds[r * 16 + k]; Q += redq[r * 16 + k]; }
            float mean = S * (1.0f / 1024.0f);
            float var  = Q * (1.0f / 1024.0f) - mean * mean;
            stm[r] = mean; stv[r] = rsqrtf(var + EPSL);
        }
        __syncthreads();
        float mean = stm[r], rq = stv[r];
        for (int c = part * 64; c < part * 64 + 64; ++c) {
            float v = (h1[r * S1 + c] - mean) * rq * g1[c] + b1[c];
            v = (v > 0.f) ? v : NEGS * v;
            h1b[r * S1 + c] = f2bf(v);
        }
    }
    __syncthreads();

    // ---- GEMM2: (16x1024) @ (1024x256); wave w owns 2 output tiles
    // 32 K-steps, A (LDS) and B (global) double-buffered, dual accumulators.
    {
        for (int t = 0; t < 2; ++t) {
            int nt = wave * 2 + t;
            v8f acc0 = {0.f, 0.f, 0.f, 0.f, 0.f, 0.f, 0.f, 0.f};
            v8f acc1 = acc0;
            v16bf aA = a_frag_lds_bf(h1b, S1, 0, lane);
            v16bf bA = b_frag_glb(w2l, 4 * CH, nt * 16, 0, lane);
            v16bf aB = a_frag_lds_bf(h1b, S1, 32, lane);
            v16bf bB = b_frag_glb(w2l, 4 * CH, nt * 16, 32, lane);
#pragma unroll
            for (int kk = 0; kk < 32; kk += 2) {
                v16bf aC, bC, aD, bD;
                if (kk + 2 < 32) {
                    aC = a_frag_lds_bf(h1b, S1, (kk + 2) * 32, lane);
                    bC = b_frag_glb(w2l, 4 * CH, nt * 16, (kk + 2) * 32, lane);
                }
                if (kk + 3 < 32) {
                    aD = a_frag_lds_bf(h1b, S1, (kk + 3) * 32, lane);
                    bD = b_frag_glb(w2l, 4 * CH, nt * 16, (kk + 3) * 32, lane);
                }
                acc0 = wmma_bf16(aA, bA, acc0);
                acc1 = wmma_bf16(aB, bB, acc1);
                aA = aC; bA = bC; aB = aD; bB = bD;
            }
            v8f acc = acc0 + acc1;
            int col = nt * 16 + nl;
#pragma unroll
            for (int i = 0; i < 8; ++i) h2[(i + half * 8) * S2 + col] = acc[i];
        }
    }
    __syncthreads();

    // ---- LN(256), relu(h + x), store f32 + bf16
    {
        int r = tid >> 4, part = tid & 15;
        float s = 0.f, q = 0.f;
        for (int c = part * 16; c < part * 16 + 16; ++c) {
            float v = h2[r * S2 + c]; s += v; q += v * v;
        }
        reds[r * 16 + part] = s; redq[r * 16 + part] = q;
        __syncthreads();
        if (part == 0) {
            float S = 0.f, Q = 0.f;
            for (int k = 0; k < 16; ++k) { S += reds[r * 16 + k]; Q += redq[r * 16 + k]; }
            float mean = S * (1.0f / CH);
            float var  = Q * (1.0f / CH) - mean * mean;
            stm[r] = mean; stv[r] = rsqrtf(var + EPSL);
        }
        __syncthreads();
        float mean = stm[r], rq = stv[r];
        size_t grow = (size_t)(row0 + r) * CH;
        for (int c = part * 16; c < part * 16 + 16; ++c) {
            float v = (h2[r * S2 + c] - mean) * rq * g2[c] + b2[c];
            float o = v + xcur[grow + c];
            o = (o > 0.f) ? o : 0.f;
            hout[grow + c]  = o;
            houtb[grow + c] = f2bf(o);
        }
    }
}

// ---------------- Kernel C: final projection out = h @ proj_w + proj_b ----------------
__global__ void __launch_bounds__(256, 1)
k_proj(const unsigned short* __restrict__ hb, const unsigned short* __restrict__ projt,
       const float* __restrict__ pbias, float* __restrict__ out) {
    const int tid = threadIdx.x, wave = tid >> 5, lane = tid & 31;
    const int row0 = blockIdx.x * 16;
    const int half = lane >> 4, nl = lane & 15;

    const unsigned short* ab = hb + (size_t)row0 * CH;
    v16bf af[8];
#pragma unroll
    for (int kk = 0; kk < 8; ++kk) af[kk] = a_frag_glb_bf(ab, CH, kk * 32, lane);

    for (int t = 0; t < 2; ++t) {
        int nt = wave * 2 + t;
        v8f acc = gemm_k8(af, projt, CH, nt * 16, lane);
        int col = nt * 16 + nl;
        float pb = pbias[col];
#pragma unroll
        for (int i = 0; i < 8; ++i)
            out[(size_t)(row0 + i + half * 8) * CH + col] = acc[i] + pb;
    }
}

// ---------------- host launcher ----------------
extern "C" void kernel_launch(void* const* d_in, const int* in_sizes, int n_in,
                              void* d_out, int out_size, void* d_ws, size_t ws_size,
                              hipStream_t stream) {
    const float* p      = (const float*)d_in[0];
    const float* x      = (const float*)d_in[1];
    const float* nrm    = (const float*)d_in[2];
    const int*   idx    = (const int*)d_in[3];
    const float* rr     = (const float*)d_in[4];
    const float* emb_w1 = (const float*)d_in[5];
    const float* emb_g1 = (const float*)d_in[6];
    const float* emb_b1 = (const float*)d_in[7];
    const float* emb_w2 = (const float*)d_in[8];
    const float* conv_w = (const float*)d_in[9];
    const float* pw_w1  = (const float*)d_in[10];
    const float* pw_g1  = (const float*)d_in[11];
    const float* pw_b1  = (const float*)d_in[12];
    const float* pw_w2  = (const float*)d_in[13];
    const float* pw_g2  = (const float*)d_in[14];
    const float* pw_b2  = (const float*)d_in[15];
    const float* proj_w = (const float*)d_in[16];
    const float* proj_b = (const float*)d_in[17];

    // workspace carve (byte offsets, 256-aligned sections)
    char* ws = (char*)d_ws;
    size_t o = 0;
    unsigned short* w2t   = (unsigned short*)(ws + o); o += (size_t)NLAY * CH * CH * 2;
    unsigned short* convb = (unsigned short*)(ws + o); o += (size_t)NLAY * 64 * 64 * 2;
    unsigned short* w1t   = (unsigned short*)(ws + o); o += (size_t)NLAY * 4 * CH * CH * 2;
    unsigned short* w2pt  = (unsigned short*)(ws + o); o += (size_t)NLAY * 4 * CH * CH * 2;
    unsigned short* projt = (unsigned short*)(ws + o); o += (size_t)CH * CH * 2;
    o = (o + 255) & ~(size_t)255;
    unsigned short* pooledb = (unsigned short*)(ws + o); o += (size_t)NPTS * CH * 2;
    float*          hA      = (float*)(ws + o);          o += (size_t)NPTS * CH * 4;
    unsigned short* hbf     = (unsigned short*)(ws + o); o += (size_t)NPTS * CH * 2;

    // ---- weight prep (bf16, B^T layouts)
    {
        int thr = 256;
        for (int l = 0; l < NLAY; ++l) {
            k_transpose_bf16<<<(CH * CH + thr - 1) / thr, thr, 0, stream>>>(
                emb_w2 + (size_t)l * CH * CH, w2t + (size_t)l * CH * CH, CH, CH);
            k_transpose_bf16<<<(CH * 4 * CH + thr - 1) / thr, thr, 0, stream>>>(
                pw_w1 + (size_t)l * CH * 4 * CH, w1t + (size_t)l * 4 * CH * CH, CH, 4 * CH);
            k_transpose_bf16<<<(4 * CH * CH + thr - 1) / thr, thr, 0, stream>>>(
                pw_w2 + (size_t)l * 4 * CH * CH, w2pt + (size_t)l * CH * 4 * CH, 4 * CH, CH);
        }
        k_convert_bf16<<<(NLAY * 64 * 64 + thr - 1) / thr, thr, 0, stream>>>(
            conv_w, convb, NLAY * 64 * 64);
        k_transpose_bf16<<<(CH * CH + thr - 1) / thr, thr, 0, stream>>>(proj_w, projt, CH, CH);
    }

    // allow >64KB dynamic LDS (WGP has 320KB)
    (void)hipFuncSetAttribute(reinterpret_cast<const void*>(k_node_embed),
                              hipFuncAttributeMaxDynamicSharedMemorySize, SMEM_KA);
    (void)hipFuncSetAttribute(reinterpret_cast<const void*>(k_pointwise),
                              hipFuncAttributeMaxDynamicSharedMemorySize, SMEM_PW);

    const float* xc = x;
    for (int l = 0; l < NLAY; ++l) {
        k_node_embed<<<NPTS / 8, 256, SMEM_KA, stream>>>(
            p, nrm, idx, xc, rr, emb_w1, emb_g1, emb_b1, w2t, convb, pooledb, l);
        float* hout = (l == 0) ? hA : (float*)d_out;  // last fp32 intermediate parked in d_out
        k_pointwise<<<NPTS / 16, 256, SMEM_PW, stream>>>(
            pooledb, xc, w1t, pw_g1, pw_b1, w2pt, pw_g2, pw_b2, hout, hbf, l);
        xc = hout;
    }
    // final projection reads only the bf16 copy, so overwriting d_out is safe
    k_proj<<<NPTS / 16, 256, 0, stream>>>(hbf, projt, proj_b, (float*)d_out);

    (void)in_sizes; (void)n_in; (void)out_size; (void)ws_size;
}
